// TorchCausalGroupedQueryPageAttentionHandler_29729763623364
// MI455X (gfx1250) — compile-verified, matrix operand order checked by reading, not run on hardware
//
// CDNA5 (gfx1250, MI455X) paged causal GQA prefill attention.
// FlashAttention-style with BOTH GEMMs transposed on the matrix pipe:
//   S^T = K · Q^T   and   O^T = V^T · P^T   (v_wmma_f32_16x16x32_bf16).
// Transposed layout keeps softmax state per-lane scalar, builds the P
// operand lane-locally from the S^T accumulators, and stores O as b128.
// KV tiles (32 positions = 2 pages) are software-pipelined through
// ping-pong LDS buffers: prefetch t+1 into registers while computing t.
// Prefetch addressing = scalar page base (SALU) + hoisted per-lane offset.
// Q is pre-scaled by sm_scale*log2e at load; Q loads / O stores are
// non-temporal so the L2 (192 MB) keeps the reused KV pool resident.
#include <hip/hip_runtime.h>

typedef __attribute__((ext_vector_type(16))) __bf16 v16bf;
typedef __attribute__((ext_vector_type(8)))  float  v8f;
typedef __attribute__((ext_vector_type(4)))  float  v4f;

#define S_   4
#define L_   1024
#define BS_  16
#define BPS_ 64
#define HQ_  32
#define HKV_ 8
#define G_   4
#define D_   128
// sm_scale * log2(e): exp(z*s - m') == exp2(z*s*log2e - m)
#define CEXP_ (0.08838834764831845f * 1.4426950408889634f)

#define LDSK_SZ (32 * 64)    // [kvpos][d/2] bf16x2 per buffer (dwords)
#define LDSV_SZ (128 * 32)   // [d][kvpos]   bf16   per buffer (ushorts)
#define PAGE_ELEMS (BS_ * HKV_ * D_)   // 16384 floats per page

union FragBF { v16bf v; unsigned u[8]; };

struct Stage { v4f k[4]; v4f v[4]; };

__device__ __forceinline__ unsigned pack2_bf16(float x, float y) {
  unsigned a = __float_as_uint(x);
  unsigned b = __float_as_uint(y);
  a = (a + 0x7FFFu + ((a >> 16) & 1u)) >> 16;   // round-to-nearest-even
  b = (b + 0x7FFFu + ((b >> 16) & 1u)) >> 16;
  return a | (b << 16);
}
__device__ __forceinline__ unsigned short bf1(float x) {
  unsigned a = __float_as_uint(x);
  return (unsigned short)((a + 0x7FFFu + ((a >> 16) & 1u)) >> 16);
}

// Starting K index of the bf16 pair in packed slot v for this lane half
// (ISA 16-bit A-matrix 16x32 layout; B mirrored with lane = N).
__device__ __forceinline__ int kpair0(int v, int ko) {
  return ((v < 4) ? (2 * v) : (16 + 2 * (v - 4))) + ko;
}

// Issue global loads for one 32-position KV tile (2 pages) into registers.
// Page select is uniform per unrolled step (it<2 -> page0); bases are SGPRs.
__device__ __forceinline__ void prefetch_tile(Stage& st,
                                              const float* __restrict__ kc,
                                              const float* __restrict__ vc,
                                              const int* __restrict__ bt,
                                              const int (&inv)[4],
                                              int s, int t) {
  const int pb0 = bt[s * BPS_ + 2 * t];
  const int pb1 = bt[s * BPS_ + 2 * t + 1];
  const float* k0 = kc + (size_t)pb0 * PAGE_ELEMS;   // scalar page bases
  const float* k1 = kc + (size_t)pb1 * PAGE_ELEMS;
  const float* v0 = vc + (size_t)pb0 * PAGE_ELEMS;
  const float* v1 = vc + (size_t)pb1 * PAGE_ELEMS;
#pragma unroll
  for (int it = 0; it < 4; ++it) {
    const float* kp = (it < 2) ? k0 : k1;   // compile-time select
    const float* vp = (it < 2) ? v0 : v1;
    st.k[it] = *(const v4f*)(kp + inv[it]); // KV stays RT: L2-resident reuse
    st.v[it] = *(const v4f*)(vp + inv[it]);
  }
}

// Convert staged registers to bf16 and commit into an LDS buffer.
__device__ __forceinline__ void commit_tile(const Stage& st,
                                            const int (&koff)[4],
                                            const int (&voff)[4],
                                            unsigned* ldsK,
                                            unsigned short* ldsVT) {
#pragma unroll
  for (int it = 0; it < 4; ++it) {
    uint2 kp = make_uint2(pack2_bf16(st.k[it].x, st.k[it].y),
                          pack2_bf16(st.k[it].z, st.k[it].w));
    *(uint2*)&ldsK[koff[it]] = kp;                   // ds_store_b64
    unsigned short* vb = ldsVT + voff[it];
    vb[0]  = bf1(st.v[it].x);                        // transposed V,
    vb[32] = bf1(st.v[it].y);                        // stride 32 -> ds offsets
    vb[64] = bf1(st.v[it].z);
    vb[96] = bf1(st.v[it].w);
  }
}

template <bool MASKED>
__device__ __forceinline__ void attn_tile(const FragBF (&qa)[4],
                                          const unsigned* ldsK,
                                          const unsigned* vt32,
                                          int q0, int kvbase,
                                          int row, int ko,
                                          v8f (&o)[8], float& m, float& l) {
  const v8f vzero = {0.f, 0.f, 0.f, 0.f, 0.f, 0.f, 0.f, 0.f};

  // ---- S^T = K · Q^T, two 16-row kv halves, f32 accumulate ----
  // A = K-tile rows (lane = kv row within half), B = Q (lane = q row).
  v8f s0 = vzero, s1 = vzero;
#pragma unroll
  for (int kk = 0; kk < 4; ++kk) {
    FragBF b0, b1;
#pragma unroll
    for (int v = 0; v < 8; ++v) {
      int k0 = kk * 32 + kpair0(v, ko);
      b0.u[v] = ldsK[(row     ) * 64 + (k0 >> 1)];
      b1.u[v] = ldsK[(row + 16) * 64 + (k0 >> 1)];
    }
    s0 = __builtin_amdgcn_wmma_f32_16x16x32_bf16(false, b0.v, false, qa[kk].v,
                                                 (short)0, s0, false, false);
    s1 = __builtin_amdgcn_wmma_f32_16x16x32_bf16(false, b1.v, false, qa[kk].v,
                                                 (short)0, s1, false, false);
  }

  // ---- per-lane softmax (Q was pre-scaled: scores already in exp2 domain)
  const int i = q0 + row;               // this lane's q row
  float z0[8], z1[8];
#pragma unroll
  for (int r = 0; r < 8; ++r) {
    z0[r] = s0[r];
    z1[r] = s1[r];
    if (MASKED) {
      const int j = kvbase + r + ko;
      if (j      > i) z0[r] = -1e30f;   // causal: mask j > i
      if (j + 16 > i) z1[r] = -1e30f;
    }
  }
  float tm = fmaxf(z0[0], z1[0]);
#pragma unroll
  for (int r = 1; r < 8; ++r) tm = fmaxf(tm, fmaxf(z0[r], z1[r]));
  tm = fmaxf(tm, __shfl_xor(tm, 16, 32));   // partner lane: other 16 kv cols
  const float mn = fmaxf(m, tm);
  const float alpha = __builtin_amdgcn_exp2f(m - mn);
  m = mn;

  float p0[8], p1[8], rs = 0.f;
#pragma unroll
  for (int r = 0; r < 8; ++r) {
    p0[r] = __builtin_amdgcn_exp2f(z0[r] - mn);
    p1[r] = __builtin_amdgcn_exp2f(z1[r] - mn);
    rs += p0[r] + p1[r];
  }
  rs += __shfl_xor(rs, 16, 32);
  l = alpha * l + rs;

  // ---- P^T operand is lane-local: pairs of consecutive kv values ----
  FragBF pa;
#pragma unroll
  for (int v = 0; v < 4; ++v) {
    pa.u[v]     = pack2_bf16(p0[2 * v], p0[2 * v + 1]);
    pa.u[v + 4] = pack2_bf16(p1[2 * v], p1[2 * v + 1]);
  }

  // ---- O^T = alpha*O^T + V^T · P^T over 8 D-tiles ----
#pragma unroll
  for (int dt = 0; dt < 8; ++dt) {
    FragBF vb;
    const int d = dt * 16 + row;        // A = V^T: lane = head-dim row
#pragma unroll
    for (int v = 0; v < 8; ++v) {
      int k0 = kpair0(v, ko);           // K = kv position, contiguous in V^T
      vb.u[v] = vt32[d * 16 + (k0 >> 1)];
    }
    v8f acc = o[dt];
#pragma unroll
    for (int r = 0; r < 8; ++r) acc[r] *= alpha;   // per-lane scalar rescale
    o[dt] = __builtin_amdgcn_wmma_f32_16x16x32_bf16(false, vb.v, false, pa.v,
                                                    (short)0, acc, false, false);
  }
}

__global__ __launch_bounds__(256)
void paged_causal_gqa_fwd(const float* __restrict__ q,
                          const float* __restrict__ kc,
                          const float* __restrict__ vc,
                          const int*   __restrict__ bt,
                          float* __restrict__ out) {
  __shared__ __align__(16) unsigned       ldsK[2 * LDSK_SZ];  // ping-pong
  __shared__ __align__(16) unsigned short ldsVT[2 * LDSV_SZ]; // ping-pong

  const int tid  = threadIdx.x;
  const int lane = tid & 31;
  const int w    = tid >> 5;            // wave 0..7
  const int g    = w >> 1;              // head-in-group 0..3
  const int qsub = w & 1;

  const int bq   = blockIdx.x & 31;         // 32-row q chunk
  const int kvh  = (blockIdx.x >> 5) & 7;   // kv head
  const int s    = blockIdx.x >> 8;         // sequence

  const int h    = kvh * G_ + g;            // query head
  const int q0   = bq * 32 + qsub * 16;     // first q row of this wave
  const int row  = lane & 15;
  const int ko   = (lane & 16) ? 8 : 0;     // lane-half offset (frag layouts)

  // Hoisted per-lane staging offsets (loop-invariant).
  int inv[4], koff[4], voff[4];
#pragma unroll
  for (int it = 0; it < 4; ++it) {
    const int f   = tid + it * 256;      // 0..1023 float4 slots
    const int pos = f >> 5;              // kv position in tile
    const int c4  = f & 31;              // float4 column (d/4)
    inv[it]  = ((pos & 15) * HKV_ + kvh) * D_ + c4 * 4;  // in-page offset
    koff[it] = pos * 64 + c4 * 2;                        // ldsK dword index
    voff[it] = c4 * 128 + pos;                           // ldsVT ushort index
  }

  // ---- Q fragments, pre-scaled by sm_scale*log2e (B operand: lane = q row)
  FragBF qa[4];
  {
    const float* qrow = q + ((size_t)((s * L_ + q0 + row) * HQ_ + h)) * D_;
#pragma unroll
    for (int kk = 0; kk < 4; ++kk) {
      const float* base = qrow + kk * 32 + ko;
      v4f f0 = __builtin_nontemporal_load((const v4f*)(base));
      v4f f1 = __builtin_nontemporal_load((const v4f*)(base + 4));
      v4f f2 = __builtin_nontemporal_load((const v4f*)(base + 16));
      v4f f3 = __builtin_nontemporal_load((const v4f*)(base + 20));
      qa[kk].u[0] = pack2_bf16(f0.x * CEXP_, f0.y * CEXP_);
      qa[kk].u[1] = pack2_bf16(f0.z * CEXP_, f0.w * CEXP_);
      qa[kk].u[2] = pack2_bf16(f1.x * CEXP_, f1.y * CEXP_);
      qa[kk].u[3] = pack2_bf16(f1.z * CEXP_, f1.w * CEXP_);
      qa[kk].u[4] = pack2_bf16(f2.x * CEXP_, f2.y * CEXP_);
      qa[kk].u[5] = pack2_bf16(f2.z * CEXP_, f2.w * CEXP_);
      qa[kk].u[6] = pack2_bf16(f3.x * CEXP_, f3.y * CEXP_);
      qa[kk].u[7] = pack2_bf16(f3.z * CEXP_, f3.w * CEXP_);
    }
  }

  const v8f vzero = {0.f, 0.f, 0.f, 0.f, 0.f, 0.f, 0.f, 0.f};
  v8f o[8];
#pragma unroll
  for (int dt = 0; dt < 8; ++dt) o[dt] = vzero;
  float m = -1e30f, l = 0.f;

  // ---- software-pipelined KV loop with ping-pong LDS buffers ----
  Stage st;
  prefetch_tile(st, kc, vc, bt, inv, s, 0);
  for (int t = 0; t < bq; ++t) {
    unsigned*       kbuf = ldsK  + (t & 1) * LDSK_SZ;
    unsigned short* vbuf = ldsVT + (t & 1) * LDSV_SZ;
    commit_tile(st, koff, voff, kbuf, vbuf);
    __syncthreads();
    prefetch_tile(st, kc, vc, bt, inv, s, t + 1);   // overlap with compute
    attn_tile<false>(qa, kbuf, (const unsigned*)vbuf, q0, t * 32,
                     row, ko, o, m, l);
  }
  {
    // Peeled diagonal tile: the only one needing the causal mask (uniform).
    unsigned*       kbuf = ldsK  + (bq & 1) * LDSK_SZ;
    unsigned short* vbuf = ldsVT + (bq & 1) * LDSV_SZ;
    commit_tile(st, koff, voff, kbuf, vbuf);
    __syncthreads();
    attn_tile<true>(qa, kbuf, (const unsigned*)vbuf, q0, bq * 32,
                    row, ko, o, m, l);
  }

  // ---- normalize and store O^T: per lane, row (q0+row), 8 consecutive
  //      columns per D-tile -> two non-temporal b128 stores each ----
  const float inv_l = __builtin_amdgcn_rcpf(l);
  float* orow = out + (size_t)(s * L_ + q0 + row) * (HQ_ * D_) + h * D_ + ko;
#pragma unroll
  for (int dt = 0; dt < 8; ++dt) {
    v4f lo = {o[dt][0] * inv_l, o[dt][1] * inv_l,
              o[dt][2] * inv_l, o[dt][3] * inv_l};
    v4f hi = {o[dt][4] * inv_l, o[dt][5] * inv_l,
              o[dt][6] * inv_l, o[dt][7] * inv_l};
    __builtin_nontemporal_store(lo, (v4f*)(orow + dt * 16));
    __builtin_nontemporal_store(hi, (v4f*)(orow + dt * 16 + 4));
  }
}

extern "C" void kernel_launch(void* const* d_in, const int* in_sizes, int n_in,
                              void* d_out, int out_size, void* d_ws, size_t ws_size,
                              hipStream_t stream) {
  const float* q  = (const float*)d_in[0];
  const float* kc = (const float*)d_in[1];
  const float* vc = (const float*)d_in[2];
  const int*   bt = (const int*)d_in[3];
  float* out = (float*)d_out;
  (void)in_sizes; (void)n_in; (void)out_size; (void)d_ws; (void)ws_size;
  dim3 grid(S_ * HKV_ * (L_ / 32));   // 1024 blocks
  paged_causal_gqa_fwd<<<grid, 256, 0, stream>>>(q, kc, vc, bt, out);
}
